// GraphAttnLayer_79602923864546
// MI455X (gfx1250) — matrix operand order repaired and checked by reading, not compiled, not run on hardware
//
#include <hip/hip_runtime.h>
#include <hip/hip_bf16.h>

typedef __attribute__((ext_vector_type(16))) _Float16 v16h;
typedef __attribute__((ext_vector_type(8)))  _Float16 v8h;
typedef __attribute__((ext_vector_type(8)))  float    v8f;
typedef __attribute__((ext_vector_type(4)))  float    f4;
typedef __attribute__((ext_vector_type(4)))  int      i4;

#define MROWS 2048
#define FIN   128
#define FOUT  128
#define NEG_INF_F (-9.0e15f)

// ---------------------------------------------------------------------------
// Kernel 1: h = x @ W  (WMMA f16->f32), writes h^T (f16) + f1 = h@a1, f2 = h@a2
// Grid: B * (M/64) = 256 blocks, 128 threads (4 waves). Wave w: rows w*16..w*16+15.
// ---------------------------------------------------------------------------
__global__ __launch_bounds__(128) void gat_h_kernel(
    const float* __restrict__ x, const float* __restrict__ W,
    const float* __restrict__ a, _Float16* __restrict__ ht,
    float* __restrict__ f1, float* __restrict__ f2)
{
  extern __shared__ char smem[];
  _Float16* Wt  = (_Float16*)smem;                      // [128][136] halves (padded)
  float*    hb  = (float*)(smem + 128 * 136 * 2);       // [64][132] f32 (padded)
  float*    aS  = (float*)(smem + 128 * 136 * 2 + 64 * 132 * 4); // [256]

  const int tid    = threadIdx.x;
  const int lane   = tid & 31;
  const int wave   = tid >> 5;
  const int b      = blockIdx.x >> 5;           // 32 blocks per batch
  const int rowBase = (blockIdx.x & 31) * 64;

  // Stage W transposed (f16) + a into LDS (coalesced over n = tid)
  for (int k = 0; k < 128; ++k)
    Wt[tid * 136 + k] = (_Float16)W[k * 128 + tid];
  aS[tid]       = a[tid];
  aS[tid + 128] = a[tid + 128];
  __syncthreads();

  const int m  = lane & 15;
  const int hi = lane >> 4;       // 0: lanes 0-15, 1: lanes 16-31
  const int kA = hi * 8;          // A-fragment k group base
  const int kB = hi * 8;          // B-fragment k group base
  const int row = rowBase + wave * 16 + m;
  const float* xrow = x + ((size_t)b * MROWS + row) * FIN;

  v8f acc[8] = {};
#pragma unroll
  for (int kc = 0; kc < 4; ++kc) {
    const int k0 = kc * 32 + kA;
    f4 xa0 = *(const f4*)(xrow + k0);
    f4 xa1 = *(const f4*)(xrow + k0 + 4);
    f4 xb0 = *(const f4*)(xrow + k0 + 16);
    f4 xb1 = *(const f4*)(xrow + k0 + 20);

    // Load ALL 8 B-fragments into distinct registers first so the 8 WMMAs
    // can issue as a dense group instead of load->wait->wmma serialization.
    v16h bf[8];
#pragma unroll
    for (int nt = 0; nt < 8; ++nt) {
      const _Float16* wp = Wt + (nt * 16 + m) * 136 + kc * 32 + kB;
      v8h b0 = *(const v8h*)wp;
      v8h b1 = *(const v8h*)(wp + 16);
#pragma unroll
      for (int e = 0; e < 8; ++e) { bf[nt][e] = b0[e]; bf[nt][8 + e] = b1[e]; }
    }

    v16h af;
#pragma unroll
    for (int e = 0; e < 4; ++e) {
      af[e]      = (_Float16)xa0[e];
      af[4 + e]  = (_Float16)xa1[e];
      af[8 + e]  = (_Float16)xb0[e];
      af[12 + e] = (_Float16)xb1[e];
    }
#pragma unroll
    for (int nt = 0; nt < 8; ++nt)
      acc[nt] = __builtin_amdgcn_wmma_f32_16x16x32_f16(
          false, af, false, bf[nt], (short)0, acc[nt], false, false);
  }

  // D layout: lane col n = nt*16 + m, rows r + hi*8
  const int rOff = wave * 16 + hi * 8;
#pragma unroll
  for (int nt = 0; nt < 8; ++nt)
#pragma unroll
    for (int r = 0; r < 8; ++r)
      hb[(rOff + r) * 132 + nt * 16 + m] = acc[nt][r];
  __syncthreads();

  // h^T (f16): thread tid = feature n, 64 contiguous rows
  {
    const int n = tid;
    _Float16* dst = ht + ((size_t)b * FOUT + n) * MROWS + rowBase;
#pragma unroll
    for (int c = 0; c < 8; ++c) {
      v8h v;
#pragma unroll
      for (int e = 0; e < 8; ++e) v[e] = (_Float16)hb[(c * 8 + e) * 132 + n];
      *(v8h*)(dst + c * 8) = v;
    }
  }
  // f1 / f2: threads 0-63 -> f1 rows, 64-127 -> f2 rows
  {
    const int r = tid & 63;
    const float* av = (tid < 64) ? aS : (aS + 128);
    float s = 0.f;
#pragma unroll 8
    for (int k = 0; k < 128; ++k) s += hb[r * 132 + k] * av[k];
    float* dst = (tid < 64) ? f1 : f2;
    dst[(size_t)b * MROWS + rowBase + r] = s;
  }
}

// ---------------------------------------------------------------------------
// Kernel 2: fused masked-softmax attention, online softmax, P@h via WMMA.
// Grid: B * (M/16) / 4 = 256 blocks, 128 threads (4 waves); 1 wave = 16 rows.
// ---------------------------------------------------------------------------
__global__ __launch_bounds__(128) void gat_attn_kernel(
    const int* __restrict__ adj, const _Float16* __restrict__ ht,
    const float* __restrict__ f1, const float* __restrict__ f2,
    float* __restrict__ out)
{
  extern __shared__ char smem[];
  const int tid  = threadIdx.x;
  const int lane = tid & 31;
  const int wave = tid >> 5;
  float* bc   = (float*)smem + wave * 16;                    // per-wave broadcast [16]
  float* obuf = (float*)(smem + 256) + wave * (16 * 128);    // per-wave [16][128]

  const int b    = blockIdx.x >> 5;                // 32 blocks per batch
  const int tile = (blockIdx.x & 31) * 4 + wave;   // 128 tiles per batch
  const int i0   = tile * 16;
  const int m  = lane & 15;
  const int hi = lane >> 4;
  const int kA = hi * 8;     // adj/e/P k group base (A-fragment)
  const int kB = hi * 8;     // h^T k group base (B-fragment)

  const int*      adjRow = adj + ((size_t)b * MROWS + (i0 + m)) * MROWS;
  const float     f1r    = f1[(size_t)b * MROWS + i0 + m];
  const float*    f2B    = f2 + (size_t)b * MROWS;
  const _Float16* htB    = ht + (size_t)b * FOUT * MROWS;

  v8f acc[8] = {};
  float rowMax = -3.0e38f;
  float rowSum = 0.f;

  for (int jb = 0; jb < MROWS; jb += 32) {
    // --- issue all global loads for this chunk up front -------------------
    i4 ad0 = *(const i4*)(adjRow + jb + kA);
    i4 ad1 = *(const i4*)(adjRow + jb + kA + 4);
    i4 ad2 = *(const i4*)(adjRow + jb + kA + 16);
    i4 ad3 = *(const i4*)(adjRow + jb + kA + 20);
    f4 g0 = *(const f4*)(f2B + jb + kA);
    f4 g1 = *(const f4*)(f2B + jb + kA + 4);
    f4 g2 = *(const f4*)(f2B + jb + kA + 16);
    f4 g3 = *(const f4*)(f2B + jb + kA + 20);

    v16h hf[8];
#pragma unroll
    for (int nt = 0; nt < 8; ++nt) {
      const _Float16* hp = htB + (size_t)(nt * 16 + m) * MROWS + jb + kB;
      v8h h0 = *(const v8h*)hp;
      v8h h1 = *(const v8h*)(hp + 16);
#pragma unroll
      for (int e = 0; e < 8; ++e) { hf[nt][e] = h0[e]; hf[nt][8 + e] = h1[e]; }
    }

    // prefetch the adj stream ~1KB ahead (the only true HBM stream)
    __builtin_prefetch(adjRow + jb + 256, 0, 1);

    // --- masked LeakyReLU scores -----------------------------------------
    float ev[16];
    {
      const f4 gs[4]  = {g0, g1, g2, g3};
      const i4 as_[4] = {ad0, ad1, ad2, ad3};
#pragma unroll
      for (int q = 0; q < 4; ++q)
#pragma unroll
        for (int e = 0; e < 4; ++e) {
          float eo = f1r + gs[q][e];
          eo = fmaxf(eo, 0.2f * eo);                         // LeakyReLU(0.2)
          ev[q * 4 + e] = (as_[q][e] > 0) ? eo : NEG_INF_F;  // mask
        }
    }

    // --- online softmax: running max across the two half-lanes of each row
    float lmax = ev[0];
#pragma unroll
    for (int t = 1; t < 16; ++t) lmax = fmaxf(lmax, ev[t]);
    lmax = fmaxf(lmax, __shfl_xor(lmax, 16, 32));
    const float mNew  = fmaxf(rowMax, lmax);
    const float scale = __expf(rowMax - mNew);
    const bool needScale = __any(scale < 1.0f);
    rowMax = mNew;

    float p[16];
    float psum = 0.f;
#pragma unroll
    for (int t = 0; t < 16; ++t) { p[t] = __expf(ev[t] - mNew); psum += p[t]; }
    psum += __shfl_xor(psum, 16, 32);
    rowSum = rowSum * scale + psum;

    if (needScale) {   // wave-uniform branch: EXEC stays full around WMMA
      if (lane < 16) bc[m] = scale;
      float s[8];
#pragma unroll
      for (int r = 0; r < 8; ++r) s[r] = bc[hi * 8 + r];
#pragma unroll
      for (int nt = 0; nt < 8; ++nt)
#pragma unroll
        for (int r = 0; r < 8; ++r) acc[nt][r] *= s[r];
    }

    // --- P (f16 A-fragment) x h^T (f16 B-fragments) -> f32 accumulators --
    v16h pf;
#pragma unroll
    for (int t = 0; t < 16; ++t) pf[t] = (_Float16)p[t];
#pragma unroll
    for (int nt = 0; nt < 8; ++nt)
      acc[nt] = __builtin_amdgcn_wmma_f32_16x16x32_f16(
          false, pf, false, hf[nt], (short)0, acc[nt], false, false);
  }

  // finalize: 1/rowSum, ELU, stage to LDS, coalesced store
  if (lane < 16) bc[m] = rowSum;
  float inv[8];
#pragma unroll
  for (int r = 0; r < 8; ++r) inv[r] = 1.0f / bc[hi * 8 + r];
#pragma unroll
  for (int nt = 0; nt < 8; ++nt)
#pragma unroll
    for (int r = 0; r < 8; ++r) {
      float v = acc[nt][r] * inv[r];
      v = (v > 0.f) ? v : (__expf(v) - 1.f);                 // ELU(alpha=1)
      obuf[(hi * 8 + r) * 128 + nt * 16 + m] = v;
    }

  float* orow = out + ((size_t)b * MROWS + i0) * FOUT;
#pragma unroll
  for (int r2 = 0; r2 < 16; ++r2) {
    f4 v = *(const f4*)(obuf + r2 * 128 + lane * 4);
    *(f4*)(orow + (size_t)r2 * FOUT + lane * 4) = v;
  }
}

// ---------------------------------------------------------------------------
extern "C" void kernel_launch(void* const* d_in, const int* in_sizes, int n_in,
                              void* d_out, int out_size, void* d_ws, size_t ws_size,
                              hipStream_t stream) {
  (void)in_sizes; (void)n_in; (void)out_size; (void)ws_size;
  const float* x   = (const float*)d_in[0];
  const int*   adj = (const int*)d_in[1];
  const float* W   = (const float*)d_in[2];
  const float* a   = (const float*)d_in[3];
  float* out = (float*)d_out;

  _Float16* ht = (_Float16*)d_ws;                                  // 8*128*2048 f16 = 4 MB
  float* f1 = (float*)((char*)d_ws + (size_t)8 * FOUT * MROWS * 2);
  float* f2 = f1 + (size_t)8 * MROWS;

  const size_t lds1 = 128 * 136 * 2 + 64 * 132 * 4 + 256 * 4;      // ~69.6 KB
  const size_t lds2 = 256 + 4 * 16 * 128 * 4;                      // ~33 KB

  gat_h_kernel<<<256, 128, lds1, stream>>>(x, W, a, ht, f1, f2);
  gat_attn_kernel<<<256, 128, lds2, stream>>>(adj, ht, f1, f2, out);
}
